// SparseGroupedExpertsPyTorch_18451179504163
// MI455X (gfx1250) — compile-verified, matrix operand order checked by reading, not run on hardware
//
#include <hip/hip_runtime.h>
#include <hip/hip_bf16.h>

// ---------------- problem constants ----------------
#define N_EXPERTS 8
#define D_MODEL   1024
#define D_FF      2048
#define N_TOKENS  8192
#define TOP_K     2
#define N_PAIRS   (N_TOKENS * TOP_K)   // 16384
#define CAP       N_PAIRS              // per-expert worst-case bucket capacity

// ---------------- tiling ----------------
#define BM   256        // pair rows per workgroup tile
#define BN   64         // output columns per workgroup tile
#define BK   32         // K per WMMA step (bf16 16x16x32)
#define TPB  256        // 8 waves of 32

typedef __attribute__((ext_vector_type(16))) __bf16 v16bf;
typedef __attribute__((ext_vector_type(8)))  float  v8f;

union Frag { v16bf v; unsigned int u[8]; };
union Acc  { v8f v; float f[8]; };

__device__ __forceinline__ unsigned short f32_bf16(float f) {
    unsigned int u = __float_as_uint(f);
    u += 0x7FFFu + ((u >> 16) & 1u);          // round-to-nearest-even
    return (unsigned short)(u >> 16);
}
__device__ __forceinline__ unsigned int pack_bf16x2(float lo, float hi) {
    return (unsigned int)f32_bf16(lo) | ((unsigned int)f32_bf16(hi) << 16);
}
// K position of the packed bf16 pair in 32-bit slot v for lane-half h
// (16-bit A/B operand layout, CDNA5 ISA §7.12.2)
__device__ __forceinline__ int kpat(int v, int h) {
    return ((v < 4) ? (2 * v) : (16 + 2 * (v - 4))) + 8 * h;
}

// ---- CDNA5 async global->LDS copy (ASYNCcnt-tracked, CDNA5 ISA) ----
// INST_OFFSET is applied to BOTH the global and LDS addresses, so consecutive
// 16B chunks share one base register pair.
template <int OFF>
__device__ __forceinline__ void async_g2l_b128(unsigned lds_off, const void* g) {
    asm volatile("global_load_async_to_lds_b128 %0, %1, off offset:%2"
                 :: "v"(lds_off), "v"((unsigned long long)(size_t)g), "i"(OFF)
                 : "memory");
}
__device__ __forceinline__ void wait_async0() {
    asm volatile("s_wait_asynccnt 0" ::: "memory");
}
__device__ __forceinline__ unsigned lds_off_of(const void* p) {
    return (unsigned)(size_t)p;   // low 32 bits of generic addr = LDS offset
}

// ---------------- kernel 1: zero output + counts ----------------
__global__ void zero_kernel(float* __restrict__ out, int* __restrict__ counts) {
    size_t i = (size_t)blockIdx.x * blockDim.x + threadIdx.x;
    size_t n4 = (size_t)N_TOKENS * D_MODEL / 4;
    if (i < n4) ((float4*)out)[i] = make_float4(0.f, 0.f, 0.f, 0.f);
    if (blockIdx.x == 0 && threadIdx.x < N_EXPERTS) counts[threadIdx.x] = 0;
}

// ---------------- kernel 2: x -> bf16 (layout unchanged) ----------------
__global__ void xcvt_kernel(const float* __restrict__ x, unsigned short* __restrict__ xb) {
    size_t i = (size_t)blockIdx.x * blockDim.x + threadIdx.x;
    if (i < (size_t)N_TOKENS * D_MODEL / 4) {
        float4 v = ((const float4*)x)[i];
        uint2 p;
        p.x = pack_bf16x2(v.x, v.y);
        p.y = pack_bf16x2(v.z, v.w);
        ((uint2*)xb)[i] = p;
    }
}

// ------- kernel 3: weight convert+transpose: [E][R][C] f32 -> [E][C][R] bf16 -------
__global__ void wcvt_kernel(const float* __restrict__ src, unsigned short* __restrict__ dst,
                            int R, int C) {
    __shared__ unsigned short tile[32][33];
    const int e = blockIdx.z;
    const float* s = src + (size_t)e * R * C;
    unsigned short* d = dst + (size_t)e * C * R;
    const int c0 = blockIdx.x * 32, r0 = blockIdx.y * 32;
    const int tx = threadIdx.x, ty = threadIdx.y;    // 32 x 8
    #pragma unroll
    for (int q = 0; q < 4; ++q) {
        int r = ty + q * 8;
        tile[r][tx] = f32_bf16(s[(size_t)(r0 + r) * C + c0 + tx]);
    }
    __syncthreads();
    #pragma unroll
    for (int q = 0; q < 4; ++q) {
        int r = ty + q * 8;
        d[(size_t)(c0 + r) * R + r0 + tx] = tile[tx][r];
    }
}

// ---------------- kernel 4: route pairs into expert buckets ----------------
__global__ void route_kernel(const long long* __restrict__ eidx,
                             int* __restrict__ counts, int* __restrict__ bucket) {
    int p = blockIdx.x * blockDim.x + threadIdx.x;
    if (p >= N_PAIRS) return;
    int e = (int)eidx[p];
    int pos = atomicAdd(&counts[e], 1);
    bucket[e * CAP + pos] = p;               // pair id = token*TOP_K + k
}

// ---------------- kernel 5: hidden = silu(xb@w1) * (xb@w2), bf16 ----------------
__global__ __launch_bounds__(TPB)
void moe_h_kernel(const unsigned short* __restrict__ xb,
                  const unsigned short* __restrict__ w1t,   // [E][D_FF][D_MODEL] bf16
                  const unsigned short* __restrict__ w2t,
                  const int*  __restrict__ counts,
                  const int*  __restrict__ bucket,
                  unsigned short* __restrict__ hidden) {
    __shared__ unsigned short sA[2][BM * BK];     // 2 x 16 KB, double buffered
    __shared__ unsigned short sB1[2][BN * BK];    // 2 x 4 KB
    __shared__ unsigned short sB2[2][BN * BK];    // 2 x 4 KB
    __shared__ int sPair[BM];

    const int e = blockIdx.z;
    const int cnt = counts[e];
    const int mTile = blockIdx.y;
    if (mTile * BM >= cnt) return;
    const int n0  = blockIdx.x * BN;
    const int tid = threadIdx.x;
    const int lane = tid & 31, wave = tid >> 5;
    const int lm = lane & 15, half = lane >> 4;

    {   int pos = mTile * BM + tid;
        sPair[tid] = (pos < cnt) ? bucket[e * CAP + pos] : -1; }
    __syncthreads();
    const int myPair = sPair[tid];
    const int myTok  = (myPair >= 0) ? (myPair >> 1) : -1;

    // padding rows: zero once (both buffers); predicated async copies skip them
    if (myTok < 0) {
        uint4 z = make_uint4(0u, 0u, 0u, 0u);
        #pragma unroll
        for (int b = 0; b < 2; ++b) {
            uint4* zr = (uint4*)&sA[b][tid * BK];
            #pragma unroll
            for (int q = 0; q < 4; ++q) zr[q] = z;
        }
    }

    Acc c1[2][BN / 16], c2[2][BN / 16];
    #pragma unroll
    for (int r = 0; r < 2; ++r)
        #pragma unroll
        for (int j = 0; j < BN / 16; ++j)
            #pragma unroll
            for (int v = 0; v < 8; ++v) { c1[r][j].f[v] = 0.f; c2[r][j].f[v] = 0.f; }

    const unsigned short* w1e = w1t + (size_t)e * D_FF * D_MODEL;
    const unsigned short* w2e = w2t + (size_t)e * D_FF * D_MODEL;
    const int nn  = tid & (BN - 1);
    const int prt = tid >> 6;                        // 0..3: 16B chunk of a 64B row
    unsigned aOff[2], b1Off[2], b2Off[2];
    #pragma unroll
    for (int b = 0; b < 2; ++b) {
        aOff[b]  = lds_off_of(&sA[b][tid * BK]);
        b1Off[b] = lds_off_of(&sB1[b][nn * BK + prt * 8]);
        b2Off[b] = lds_off_of(&sB2[b][nn * BK + prt * 8]);
    }
    const unsigned short* aSrc  = (myTok >= 0) ? xb + (size_t)myTok * D_MODEL : nullptr;
    const unsigned short* b1Src = w1e + (size_t)(n0 + nn) * D_MODEL + prt * 8;
    const unsigned short* b2Src = w2e + (size_t)(n0 + nn) * D_MODEL + prt * 8;

    auto stage = [&](int b, int k0) {
        if (myTok >= 0) {
            const unsigned short* s = aSrc + k0;
            async_g2l_b128<0 >(aOff[b], s);
            async_g2l_b128<16>(aOff[b], s);
            async_g2l_b128<32>(aOff[b], s);
            async_g2l_b128<48>(aOff[b], s);
        }
        async_g2l_b128<0>(b1Off[b], b1Src + k0);
        async_g2l_b128<0>(b2Off[b], b2Src + k0);
    };

    const int S = D_MODEL / BK;
    stage(0, 0);
    wait_async0();
    __syncthreads();

    for (int s = 0; s < S; ++s) {
        if (s + 1 < S) stage((s + 1) & 1, (s + 1) * BK);  // overlap next copy w/ compute
        const int cb = s & 1;

        Frag a[2], b1[BN / 16], b2[BN / 16];
        #pragma unroll
        for (int r = 0; r < 2; ++r) {
            const unsigned short* ar = &sA[cb][(wave * 32 + r * 16 + lm) * BK];
            #pragma unroll
            for (int v = 0; v < 8; ++v)
                a[r].u[v] = *(const unsigned int*)&ar[kpat(v, half)];
        }
        #pragma unroll
        for (int j = 0; j < BN / 16; ++j) {
            const unsigned short* br1 = &sB1[cb][(j * 16 + lm) * BK];
            const unsigned short* br2 = &sB2[cb][(j * 16 + lm) * BK];
            #pragma unroll
            for (int v = 0; v < 8; ++v) {
                int kb = kpat(v, half);
                b1[j].u[v] = *(const unsigned int*)&br1[kb];
                b2[j].u[v] = *(const unsigned int*)&br2[kb];
            }
        }
        #pragma unroll
        for (int j = 0; j < BN / 16; ++j)
            #pragma unroll
            for (int r = 0; r < 2; ++r)
                c1[r][j].v = __builtin_amdgcn_wmma_f32_16x16x32_bf16(
                    false, a[r].v, false, b1[j].v, (short)0, c1[r][j].v, false, false);
        #pragma unroll
        for (int j = 0; j < BN / 16; ++j)
            #pragma unroll
            for (int r = 0; r < 2; ++r)
                c2[r][j].v = __builtin_amdgcn_wmma_f32_16x16x32_bf16(
                    false, a[r].v, false, b2[j].v, (short)0, c2[r][j].v, false, false);

        wait_async0();       // next buffer's copies (this wave) complete
        __syncthreads();     // all waves done reading cb + copies visible
    }

    // ---- epilogue: SwiGLU, write hidden bf16 at pair-id rows ----
    #pragma unroll
    for (int r = 0; r < 2; ++r)
        #pragma unroll
        for (int j = 0; j < BN / 16; ++j)
            #pragma unroll
            for (int v = 0; v < 8; ++v) {
                int m = wave * 32 + r * 16 + v + 8 * half;  // C/D: M = vgpr + 8*half
                int pid = sPair[m];
                if (pid >= 0) {
                    float g = c1[r][j].f[v];
                    float sg = g / (1.f + __expf(-g));
                    float h = sg * c2[r][j].f[v];
                    hidden[(size_t)pid * D_FF + n0 + j * 16 + lm] = f32_bf16(h);
                }
            }
}

// ---------------- kernel 6: out += route_w * (hidden @ w3) ----------------
__global__ __launch_bounds__(TPB)
void moe_o_kernel(const unsigned short* __restrict__ hidden,
                  const unsigned short* __restrict__ w3t,   // [E][D_MODEL][D_FF] bf16
                  const float* __restrict__ ewgt,
                  const int*  __restrict__ counts,
                  const int*  __restrict__ bucket,
                  float* __restrict__ out) {
    __shared__ unsigned short sA[2][BM * BK];
    __shared__ unsigned short sB[2][BN * BK];
    __shared__ int sPair[BM];

    const int e = blockIdx.z;
    const int cnt = counts[e];
    const int mTile = blockIdx.y;
    if (mTile * BM >= cnt) return;
    const int n0  = blockIdx.x * BN;                 // over D_MODEL
    const int tid = threadIdx.x;
    const int lane = tid & 31, wave = tid >> 5;
    const int lm = lane & 15, half = lane >> 4;

    {   int pos = mTile * BM + tid;
        sPair[tid] = (pos < cnt) ? bucket[e * CAP + pos] : -1; }
    __syncthreads();
    const int myPair = sPair[tid];

    if (myPair < 0) {
        uint4 z = make_uint4(0u, 0u, 0u, 0u);
        #pragma unroll
        for (int b = 0; b < 2; ++b) {
            uint4* zr = (uint4*)&sA[b][tid * BK];
            #pragma unroll
            for (int q = 0; q < 4; ++q) zr[q] = z;
        }
    }

    Acc c[2][BN / 16];
    #pragma unroll
    for (int r = 0; r < 2; ++r)
        #pragma unroll
        for (int j = 0; j < BN / 16; ++j)
            #pragma unroll
            for (int v = 0; v < 8; ++v) c[r][j].f[v] = 0.f;

    const unsigned short* w3e = w3t + (size_t)e * D_MODEL * D_FF;
    const int nn  = tid & (BN - 1);
    const int prt = tid >> 6;
    unsigned aOff[2], bOff[2];
    #pragma unroll
    for (int b = 0; b < 2; ++b) {
        aOff[b] = lds_off_of(&sA[b][tid * BK]);
        bOff[b] = lds_off_of(&sB[b][nn * BK + prt * 8]);
    }
    const unsigned short* aSrc = (myPair >= 0) ? hidden + (size_t)myPair * D_FF : nullptr;
    const unsigned short* bSrc = w3e + (size_t)(n0 + nn) * D_FF + prt * 8;

    auto stage = [&](int b, int k0) {
        if (myPair >= 0) {
            const unsigned short* s = aSrc + k0;
            async_g2l_b128<0 >(aOff[b], s);
            async_g2l_b128<16>(aOff[b], s);
            async_g2l_b128<32>(aOff[b], s);
            async_g2l_b128<48>(aOff[b], s);
        }
        async_g2l_b128<0>(bOff[b], bSrc + k0);
    };

    const int S = D_FF / BK;
    stage(0, 0);
    wait_async0();
    __syncthreads();

    for (int s = 0; s < S; ++s) {
        if (s + 1 < S) stage((s + 1) & 1, (s + 1) * BK);
        const int cb = s & 1;

        Frag a[2], b[BN / 16];
        #pragma unroll
        for (int r = 0; r < 2; ++r) {
            const unsigned short* ar = &sA[cb][(wave * 32 + r * 16 + lm) * BK];
            #pragma unroll
            for (int v = 0; v < 8; ++v)
                a[r].u[v] = *(const unsigned int*)&ar[kpat(v, half)];
        }
        #pragma unroll
        for (int j = 0; j < BN / 16; ++j) {
            const unsigned short* br = &sB[cb][(j * 16 + lm) * BK];
            #pragma unroll
            for (int v = 0; v < 8; ++v)
                b[j].u[v] = *(const unsigned int*)&br[kpat(v, half)];
        }
        #pragma unroll
        for (int j = 0; j < BN / 16; ++j)
            #pragma unroll
            for (int r = 0; r < 2; ++r)
                c[r][j].v = __builtin_amdgcn_wmma_f32_16x16x32_bf16(
                    false, a[r].v, false, b[j].v, (short)0, c[r][j].v, false, false);

        wait_async0();
        __syncthreads();
    }

    // weighted scatter-add; exactly 2 commutative adds per token -> deterministic
    #pragma unroll
    for (int r = 0; r < 2; ++r)
        #pragma unroll
        for (int j = 0; j < BN / 16; ++j)
            #pragma unroll
            for (int v = 0; v < 8; ++v) {
                int m = wave * 32 + r * 16 + v + 8 * half;
                int pid = sPair[m];
                if (pid >= 0) {
                    int tok = pid >> 1;
                    float w = ewgt[pid];
                    unsafeAtomicAdd(&out[(size_t)tok * D_MODEL + n0 + j * 16 + lm],
                                    w * c[r][j].f[v]);
                }
            }
}

// ---------------- host ----------------
extern "C" void kernel_launch(void* const* d_in, const int* in_sizes, int n_in,
                              void* d_out, int out_size, void* d_ws, size_t ws_size,
                              hipStream_t stream) {
    const float*     x    = (const float*)d_in[0];
    const long long* eidx = (const long long*)d_in[1];   // int64 indices
    const float*     ewgt = (const float*)d_in[2];
    const float*     w1   = (const float*)d_in[3];
    const float*     w2   = (const float*)d_in[4];
    const float*     w3   = (const float*)d_in[5];
    float*           out  = (float*)d_out;

    char* ws = (char*)d_ws;
    const size_t WSZ = (size_t)N_EXPERTS * D_MODEL * D_FF * 2;  // 32 MB each
    int* counts = (int*)ws;                                      // 32 B
    int* bucket = (int*)(ws + 256);                              // 512 KB
    unsigned short* xb  = (unsigned short*)(ws + (1u << 20));    // 16 MB
    unsigned short* w1t = xb  + (size_t)N_TOKENS * D_MODEL;
    unsigned short* w2t = w1t + WSZ / 2;
    unsigned short* w3t = w2t + WSZ / 2;
    unsigned short* hidden = w3t + WSZ / 2;                      // 64 MB bf16

    zero_kernel<<<(N_TOKENS * D_MODEL / 4 + TPB - 1) / TPB, TPB, 0, stream>>>(out, counts);
    xcvt_kernel<<<(N_TOKENS * D_MODEL / 4 + TPB - 1) / TPB, TPB, 0, stream>>>(x, xb);

    dim3 tb(32, 8);
    dim3 g12(D_FF / 32, D_MODEL / 32, N_EXPERTS);
    wcvt_kernel<<<g12, tb, 0, stream>>>(w1, w1t, D_MODEL, D_FF);   // -> [E][D_FF][D_MODEL]
    wcvt_kernel<<<g12, tb, 0, stream>>>(w2, w2t, D_MODEL, D_FF);
    dim3 g3(D_MODEL / 32, D_FF / 32, N_EXPERTS);
    wcvt_kernel<<<g3, tb, 0, stream>>>(w3, w3t, D_FF, D_MODEL);    // -> [E][D_MODEL][D_FF]

    route_kernel<<<(N_PAIRS + TPB - 1) / TPB, TPB, 0, stream>>>(eidx, counts, bucket);

    dim3 gh(D_FF / BN,    N_PAIRS / BM, N_EXPERTS);   // (32, 64, 8)
    moe_h_kernel<<<gh, TPB, 0, stream>>>(xb, w1t, w2t, counts, bucket, hidden);

    dim3 go(D_MODEL / BN, N_PAIRS / BM, N_EXPERTS);   // (16, 64, 8)
    moe_o_kernel<<<go, TPB, 0, stream>>>(hidden, w3t, ewgt, counts, bucket, out);
}
